// DNN_89678917141217
// MI455X (gfx1250) — matrix operand order repaired and checked by reading, not compiled, not run on hardware
//
#include <hip/hip_runtime.h>
#include <hip/hip_bf16.h>
#include <stdint.h>

typedef __attribute__((ext_vector_type(16))) __bf16 v16bf;
typedef __attribute__((ext_vector_type(8)))  float  v8f;

// ---------- bf16 helpers ----------
__device__ __forceinline__ unsigned short f2bf(float f) {
    unsigned int u = __builtin_bit_cast(unsigned int, f);
    unsigned int r = u + 0x7FFFu + ((u >> 16) & 1u);   // round-to-nearest-even
    return (unsigned short)(r >> 16);
}
__device__ __forceinline__ float bf2f(unsigned short h) {
    unsigned int u = ((unsigned int)h) << 16;
    return __builtin_bit_cast(float, u);
}

union V16Cast { uint4 q[2]; v16bf v; };

// =====================================================================
// Kernel 1: embedding gather * score, mean-pool over S=128 -> bf16 [B,1024]
// grid = B (1024), block = 256, each thread owns 4 contiguous dims
// =====================================================================
__global__ __launch_bounds__(256)
void pool_kernel(const int* __restrict__ sentence, const float* __restrict__ scores,
                 const float* __restrict__ emb, unsigned short* __restrict__ x0) {
    const int b   = blockIdx.x;
    const int tid = threadIdx.x;
    __shared__ int   tok[128];
    __shared__ float sc[128];
    if (tid < 128) {
        tok[tid] = sentence[b * 128 + tid];
        sc[tid]  = scores[b * 128 + tid];
    }
    __syncthreads();

    const int d = tid * 4;                       // 256*4 = 1024 dims
    float4 acc = make_float4(0.f, 0.f, 0.f, 0.f);
    for (int s = 0; s < 128; ++s) {
        const float4 e = *reinterpret_cast<const float4*>(&emb[(size_t)tok[s] * 1024 + d]);
        const float w = sc[s];
        acc.x += e.x * w; acc.y += e.y * w; acc.z += e.z * w; acc.w += e.w * w;
    }
    const float inv = 1.0f / 128.0f;
    unsigned short* o = x0 + (size_t)b * 1024 + d;
    o[0] = f2bf(acc.x * inv); o[1] = f2bf(acc.y * inv);
    o[2] = f2bf(acc.z * inv); o[3] = f2bf(acc.w * inv);
}

// =====================================================================
// Kernel 2: O[M,N](bf16) = A[M,K](bf16) @ W[N,K](f32->bf16)^T + bias
// 128x128 C-tile per 256-thread block; 8 waves each own 32(M) x 64(N);
// v_wmma_f32_16x16x32_bf16, K-step 32.
// Double-buffered LDS + register prefetch: next tile's global_load_b128s
// issue before the current WMMA block, hiding HBM latency behind the
// matrix pipe. LDS rows padded to 40 shorts (80B stride: 16B-aligned
// b128 rows, 20-dword stride -> conflict-free across 16 rows).
// =====================================================================
#define BM 128
#define BN 128
#define BK 32
#define LDP 40

__global__ __launch_bounds__(256)
void gemm_bf16_wmma(const unsigned short* __restrict__ A,
                    const float* __restrict__ W,
                    const float* __restrict__ bias,
                    unsigned short* __restrict__ O,
                    int M, int N, int K) {
    __shared__ __align__(16) unsigned short As[2][BM][LDP];
    __shared__ __align__(16) unsigned short Ws[2][BN][LDP];

    const int tid  = threadIdx.x;
    const int lane = tid & 31;
    const int wave = tid >> 5;
    const int m0 = blockIdx.y * BM;
    const int n0 = blockIdx.x * BN;

    const int waveM = (wave & 3) * 32;   // 0,32,64,96
    const int waveN = (wave >> 2) * 64;  // 0,64
    const int half  = lane >> 4;         // lane group 0..15 / 16..31
    const int lr    = lane & 15;

    // staging coordinates (8 x ushort4/float4 per 32-elem row)
    int sr[4], scl[4];
    #pragma unroll
    for (int i = 0; i < 4; ++i) {
        const int idx = tid + i * 256;
        sr[i]  = idx >> 3;
        scl[i] = (idx & 7) * 4;
    }

    const v8f zerov = {0.f, 0.f, 0.f, 0.f, 0.f, 0.f, 0.f, 0.f};
    v8f acc[2][4];
    #pragma unroll
    for (int mi = 0; mi < 2; ++mi)
        #pragma unroll
        for (int ni = 0; ni < 4; ++ni) acc[mi][ni] = zerov;

    // ---- prologue: stage tile 0 into buffer 0 ----
    ushort4 ra[4];
    float4  rw[4];
    #pragma unroll
    for (int i = 0; i < 4; ++i) {
        ra[i] = *reinterpret_cast<const ushort4*>(&A[(size_t)(m0 + sr[i]) * K + scl[i]]);
        rw[i] = *reinterpret_cast<const float4*>(&W[(size_t)(n0 + sr[i]) * K + scl[i]]);
    }
    #pragma unroll
    for (int i = 0; i < 4; ++i) {
        *reinterpret_cast<ushort4*>(&As[0][sr[i]][scl[i]]) = ra[i];
        ushort4 h;
        h.x = f2bf(rw[i].x); h.y = f2bf(rw[i].y); h.z = f2bf(rw[i].z); h.w = f2bf(rw[i].w);
        *reinterpret_cast<ushort4*>(&Ws[0][sr[i]][scl[i]]) = h;
    }
    __syncthreads();

    const int steps = K / BK;
    for (int s = 0; s < steps; ++s) {
        const int buf = s & 1;

        // ---- prefetch next tile into registers (overlaps with WMMA) ----
        ushort4 pa[4];
        float4  pw[4];
        if (s + 1 < steps) {
            const int kn = (s + 1) * BK;
            #pragma unroll
            for (int i = 0; i < 4; ++i) {
                pa[i] = *reinterpret_cast<const ushort4*>(
                    &A[(size_t)(m0 + sr[i]) * K + kn + scl[i]]);
                pw[i] = *reinterpret_cast<const float4*>(
                    &W[(size_t)(n0 + sr[i]) * K + kn + scl[i]]);
            }
        }

        // ---- fragments from current buffer ----
        // A 16x32 bf16: lanes0-15 row=lr halves = K0..7 / K16..23
        //               lanes16-31 row=lr halves = K8..15 / K24..31
        v16bf afrag[2];
        #pragma unroll
        for (int mi = 0; mi < 2; ++mi) {
            const int row = waveM + mi * 16 + lr;
            const int kb  = half * 8;
            V16Cast u;
            u.q[0] = *reinterpret_cast<const uint4*>(&As[buf][row][kb]);
            u.q[1] = *reinterpret_cast<const uint4*>(&As[buf][row][kb + 16]);
            afrag[mi] = u.v;
        }
        // B 32x16 bf16: lanes0-15 col=lr hold K0..7 / K16..23, lanes16-31 K8..15 / K24..31
        v16bf bfrag[4];
        #pragma unroll
        for (int ni = 0; ni < 4; ++ni) {
            const int row = waveN + ni * 16 + lr;  // output-neuron row of W
            const int kb  = half * 8;
            V16Cast u;
            u.q[0] = *reinterpret_cast<const uint4*>(&Ws[buf][row][kb]);
            u.q[1] = *reinterpret_cast<const uint4*>(&Ws[buf][row][kb + 16]);
            bfrag[ni] = u.v;
        }

        #pragma unroll
        for (int mi = 0; mi < 2; ++mi)
            #pragma unroll
            for (int ni = 0; ni < 4; ++ni)
                acc[mi][ni] = __builtin_amdgcn_wmma_f32_16x16x32_bf16(
                    false, afrag[mi], false, bfrag[ni],
                    (short)0, acc[mi][ni], false, false);

        // ---- drain prefetch into the other buffer ----
        if (s + 1 < steps) {
            __syncthreads();   // readers of buf^1 (iteration s-1) are done
            #pragma unroll
            for (int i = 0; i < 4; ++i) {
                *reinterpret_cast<ushort4*>(&As[buf ^ 1][sr[i]][scl[i]]) = pa[i];
                ushort4 h;
                h.x = f2bf(pw[i].x); h.y = f2bf(pw[i].y);
                h.z = f2bf(pw[i].z); h.w = f2bf(pw[i].w);
                *reinterpret_cast<ushort4*>(&Ws[buf ^ 1][sr[i]][scl[i]]) = h;
            }
            __syncthreads();   // stores visible before next compute phase
        }
    }

    // ---- epilogue: +bias, cvt bf16, store ----
    #pragma unroll
    for (int ni = 0; ni < 4; ++ni) {
        const int col = n0 + waveN + ni * 16 + lr;
        const float bv = bias[col];
        #pragma unroll
        for (int mi = 0; mi < 2; ++mi) {
            #pragma unroll
            for (int r = 0; r < 8; ++r) {
                const int rowg = m0 + waveM + mi * 16 + half * 8 + r;
                const float val = acc[mi][ni][r] + bv;
                O[(size_t)rowg * N + col] = f2bf(val);
            }
        }
    }
}

// =====================================================================
// Kernel 3: logits = x @ Wout^T + bout ; log_softmax over 4 labels
// grid = B, block = 128 (4 waves, one label per wave)
// =====================================================================
__global__ __launch_bounds__(128)
void head_kernel(const unsigned short* __restrict__ x,
                 const float* __restrict__ Wout,
                 const float* __restrict__ bout,
                 float* __restrict__ out) {
    const int b    = blockIdx.x;
    const int lane = threadIdx.x & 31;
    const int lab  = threadIdx.x >> 5;          // 0..3

    const unsigned short* xr = x + (size_t)b * 4096;
    const float* wr = Wout + (size_t)lab * 4096;

    float acc = 0.f;
    for (int k = lane; k < 4096; k += 32)
        acc += bf2f(xr[k]) * wr[k];
    #pragma unroll
    for (int off = 16; off > 0; off >>= 1)
        acc += __shfl_down(acc, off, 32);

    __shared__ float logits[4];
    if (lane == 0) logits[lab] = acc + bout[lab];
    __syncthreads();

    if (threadIdx.x == 0) {
        float m = logits[0];
        for (int i = 1; i < 4; ++i) m = fmaxf(m, logits[i]);
        float s = 0.f;
        for (int i = 0; i < 4; ++i) s += __expf(logits[i] - m);
        const float lse = m + __logf(s);
        for (int i = 0; i < 4; ++i) out[(size_t)b * 4 + i] = logits[i] - lse;
    }
}

// =====================================================================
// Launch
// =====================================================================
extern "C" void kernel_launch(void* const* d_in, const int* in_sizes, int n_in,
                              void* d_out, int out_size, void* d_ws, size_t ws_size,
                              hipStream_t stream) {
    const int*   sentence = (const int*)  d_in[0];
    const float* scores   = (const float*)d_in[1];
    const float* emb      = (const float*)d_in[2];
    const float* W1 = (const float*)d_in[3];  const float* b1 = (const float*)d_in[4];
    const float* W2 = (const float*)d_in[5];  const float* b2 = (const float*)d_in[6];
    const float* W3 = (const float*)d_in[7];  const float* b3 = (const float*)d_in[8];
    const float* W4 = (const float*)d_in[9];  const float* b4 = (const float*)d_in[10];
    const float* W5 = (const float*)d_in[11]; const float* b5 = (const float*)d_in[12];
    const float* Wout = (const float*)d_in[13];
    const float* bout = (const float*)d_in[14];
    float* out = (float*)d_out;

    const int B = 1024, EMB = 1024, HID = 4096;

    char* ws = (char*)d_ws;
    unsigned short* x0 = (unsigned short*)(ws);                         // 2 MB  [B,EMB]
    unsigned short* x1 = (unsigned short*)(ws + (size_t)(2  << 20));    // 8 MB  [B,HID]
    unsigned short* x2 = (unsigned short*)(ws + (size_t)(10 << 20));    // 8 MB  [B,HID]

    // 1) embedding gather + mean pool -> bf16
    pool_kernel<<<B, 256, 0, stream>>>(sentence, scores, emb, x0);

    // 2) 5 dense layers via bf16 WMMA (f32 accumulate), double-buffered
    dim3 blk(256);
    dim3 grd(HID / BN, B / BM);  // (32, 8)
    gemm_bf16_wmma<<<grd, blk, 0, stream>>>(x0, W1, b1, x1, B, HID, EMB);
    gemm_bf16_wmma<<<grd, blk, 0, stream>>>(x1, W2, b2, x2, B, HID, HID);
    gemm_bf16_wmma<<<grd, blk, 0, stream>>>(x2, W3, b3, x1, B, HID, HID);
    gemm_bf16_wmma<<<grd, blk, 0, stream>>>(x1, W4, b4, x2, B, HID, HID);
    gemm_bf16_wmma<<<grd, blk, 0, stream>>>(x2, W5, b5, x1, B, HID, HID);

    // 3) head + log_softmax
    head_kernel<<<B, 128, 0, stream>>>(x1, Wout, bout, out);
}
